// CLoss_17145509446102
// MI455X (gfx1250) — matrix operand order repaired and checked by reading, not compiled, not run on hardware
//
#include <hip/hip_runtime.h>
#include <math.h>

typedef __attribute__((ext_vector_type(16))) __bf16 v16bf;
typedef __attribute__((ext_vector_type(8)))  __bf16 v8bf;
typedef __attribute__((ext_vector_type(8)))  float  v8f;

static constexpr int Nn = 4096;   // rows of feat
static constexpr int Mm = 4096;   // rows of feat2 (classes)
static constexpr int Dd = 512;    // feature dim

// ---------------------------------------------------------------------------
// Kernel 1: squared row norms of feat (-> x2) and feat2 (-> y2). One wave/row.
// ---------------------------------------------------------------------------
__global__ __launch_bounds__(256) void closs_norms(const float* __restrict__ feat,
                                                   const float* __restrict__ feat2,
                                                   float* __restrict__ x2,
                                                   float* __restrict__ y2) {
    const int lane = threadIdx.x & 31;
    const int wrow = (blockIdx.x * blockDim.x + threadIdx.x) >> 5;
    if (wrow >= Nn + Mm) return;
    const float* src = (wrow < Nn) ? (feat + (size_t)wrow * Dd)
                                   : (feat2 + (size_t)(wrow - Nn) * Dd);
    float s = 0.0f;
    #pragma unroll
    for (int k = lane * 4; k < Dd; k += 32 * 4) {
        float4 v = *(const float4*)(src + k);
        s += v.x * v.x + v.y * v.y + v.z * v.z + v.w * v.w;
    }
    #pragma unroll
    for (int off = 16; off >= 1; off >>= 1)
        s += __shfl_xor(s, off, 32);
    if (lane == 0) {
        if (wrow < Nn) x2[wrow] = s; else y2[wrow - Nn] = s;
    }
}

// ---------------------------------------------------------------------------
// Kernel 1b: split fp32 -> (bf16 hi, bf16 lo) arrays, row-major, once.
// ---------------------------------------------------------------------------
__global__ __launch_bounds__(256) void closs_split(const float* __restrict__ src,
                                                   unsigned short* __restrict__ hi,
                                                   unsigned short* __restrict__ lo,
                                                   int n4) {
    int i = blockIdx.x * blockDim.x + threadIdx.x;
    if (i >= n4) return;
    float4 v = ((const float4*)src)[i];
    float f[4] = {v.x, v.y, v.z, v.w};
    unsigned short h[4], l[4];
    #pragma unroll
    for (int j = 0; j < 4; ++j) {
        __bf16 hb = (__bf16)f[j];
        __bf16 lb = (__bf16)(f[j] - (float)hb);
        h[j] = __builtin_bit_cast(unsigned short, hb);
        l[j] = __builtin_bit_cast(unsigned short, lb);
    }
    *(ushort4*)(hi + (size_t)i * 4) = make_ushort4(h[0], h[1], h[2], h[3]);
    *(ushort4*)(lo + (size_t)i * 4) = make_ushort4(l[0], l[1], l[2], l[3]);
}

// ---------------------------------------------------------------------------
// Shared epilogue helpers
// ---------------------------------------------------------------------------
__device__ __forceinline__ void lse_update(float& m, float& s, float lg) {
    float nm = fmaxf(m, lg);
    s = s * __expf(m - nm) + __expf(lg - nm);
    m = nm;
}
__device__ __forceinline__ void lse_merge(float& m, float& s, float m2, float s2) {
    float nm = fmaxf(m, m2);
    s = s * __expf(m - nm) + s2 * __expf(m2 - nm);
    m = nm;
}

// ---------------------------------------------------------------------------
// Kernel 2 (fast path): fused distance-GEMM + online log-softmax CE.
// Pre-split bf16 hi/lo operands; each wave processes TWO 16x16 column tiles
// per iteration so every A load feeds 6 WMMAs (2.0 b128-loads per WMMA).
// Block = 256 threads (8 waves) owns 16 rows; waves stripe 128 tile-pairs.
// ---------------------------------------------------------------------------
__global__ __launch_bounds__(256) void closs_main_pre(
        const unsigned short* __restrict__ fhi, const unsigned short* __restrict__ flo,
        const unsigned short* __restrict__ ghi, const unsigned short* __restrict__ glo,
        const int*   __restrict__ labels, const float* __restrict__ temp,
        const float* __restrict__ x2, const float* __restrict__ y2,
        float* __restrict__ nll) {
    __shared__ float lds_x2[16];
    __shared__ float lds_lablogit[16];
    __shared__ int   lds_lab[16];
    __shared__ float lds_m[16][8];
    __shared__ float lds_s[16][8];

    const int tid  = threadIdx.x;
    const int lane = tid & 31;
    const int wave = tid >> 5;
    const int hl   = lane >> 4;     // half-wave: 0 or 1
    const int lid  = lane & 15;
    const int r0   = blockIdx.x * 16;

    if (tid < 16) {
        lds_x2[tid]       = x2[r0 + tid];
        lds_lab[tid]      = labels[r0 + tid];
        lds_lablogit[tid] = 0.0f;
    }
    __syncthreads();

    const float inv_temp = 1.0f / temp[0];

    // A operand (16x32 bf16 layout): lane holds row `lid`,
    // K = hl*16 + {0..15} relative to each k0 (contiguous).
    const unsigned short* ah = fhi + (size_t)(r0 + lid) * Dd + hl * 16;
    const unsigned short* al = flo + (size_t)(r0 + lid) * Dd + hl * 16;

    float mrun[8], srun[8];
    #pragma unroll
    for (int v = 0; v < 8; ++v) { mrun[v] = -INFINITY; srun[v] = 0.0f; }

    for (int tp = wave; tp < Mm / 32; tp += 8) {
        const int c0 = tp * 32;            // two adjacent 16-col tiles: c0, c0+16
        // B operand (32x16 bf16 layout): lane holds col `lid`,
        // K = hl*16 + {0..15} relative to each k0 (contiguous).
        // Second tile is a constant +16*Dd element offset (folds into ioffset).
        const unsigned short* bh = ghi + (size_t)(c0 + lid) * Dd + hl * 16;
        const unsigned short* bl = glo + (size_t)(c0 + lid) * Dd + hl * 16;

        v8f acc0a = {}, acc1a = {}, acc2a = {};
        v8f acc0b = {}, acc1b = {}, acc2b = {};
        #pragma unroll 2
        for (int k0 = 0; k0 < Dd; k0 += 32) {
            v16bf ahi, alo;
            *(v8bf*)&ahi         = *(const v8bf*)(ah + k0);
            *(((v8bf*)&ahi) + 1) = *(const v8bf*)(ah + k0 + 8);
            *(v8bf*)&alo         = *(const v8bf*)(al + k0);
            *(((v8bf*)&alo) + 1) = *(const v8bf*)(al + k0 + 8);
            v16bf bhi0 = *(const v16bf*)(bh + k0);
            v16bf blo0 = *(const v16bf*)(bl + k0);
            v16bf bhi1 = *(const v16bf*)(bh + k0 + 16 * Dd);
            v16bf blo1 = *(const v16bf*)(bl + k0 + 16 * Dd);
            // dot ~= hi*hi + hi*lo + lo*hi  (fp32 accumulate, 6 indep chains)
            acc0a = __builtin_amdgcn_wmma_f32_16x16x32_bf16(false, ahi, false, bhi0,
                                                            (short)0, acc0a, false, false);
            acc1a = __builtin_amdgcn_wmma_f32_16x16x32_bf16(false, ahi, false, blo0,
                                                            (short)0, acc1a, false, false);
            acc2a = __builtin_amdgcn_wmma_f32_16x16x32_bf16(false, alo, false, bhi0,
                                                            (short)0, acc2a, false, false);
            acc0b = __builtin_amdgcn_wmma_f32_16x16x32_bf16(false, ahi, false, bhi1,
                                                            (short)0, acc0b, false, false);
            acc1b = __builtin_amdgcn_wmma_f32_16x16x32_bf16(false, ahi, false, blo1,
                                                            (short)0, acc1b, false, false);
            acc2b = __builtin_amdgcn_wmma_f32_16x16x32_bf16(false, alo, false, bhi1,
                                                            (short)0, acc2b, false, false);
        }

        // C element (lane, v) -> row v + 8*hl, col cc + lid
        #pragma unroll
        for (int t = 0; t < 2; ++t) {
            const int cc = c0 + t * 16;
            const float y2c = y2[cc + lid];
            #pragma unroll
            for (int v = 0; v < 8; ++v) {
                const int row = v + 8 * hl;
                float dot = (t == 0) ? (acc0a[v] + (acc1a[v] + acc2a[v]))
                                     : (acc0b[v] + (acc1b[v] + acc2b[v]));
                float d2   = lds_x2[row] + y2c - 2.0f * dot;
                float dist = __builtin_amdgcn_sqrtf(fmaxf(d2, 0.0f));
                float lg   = -dist * inv_temp;
                if (lds_lab[row] == cc + lid) lds_lablogit[row] = lg;  // unique writer
                lse_update(mrun[v], srun[v], lg);
            }
        }
    }

    // Combine the 16 column-slot lanes within each half-wave.
    #pragma unroll
    for (int off = 1; off < 16; off <<= 1) {
        #pragma unroll
        for (int v = 0; v < 8; ++v) {
            float m2 = __shfl_xor(mrun[v], off, 32);
            float s2 = __shfl_xor(srun[v], off, 32);
            lse_merge(mrun[v], srun[v], m2, s2);
        }
    }
    if (lid == 0) {
        #pragma unroll
        for (int v = 0; v < 8; ++v) {
            lds_m[v + 8 * hl][wave] = mrun[v];
            lds_s[v + 8 * hl][wave] = srun[v];
        }
    }
    __syncthreads();

    if (tid < 16) {
        float m = lds_m[tid][0];
        float s = lds_s[tid][0];
        #pragma unroll
        for (int w = 1; w < 8; ++w)
            lse_merge(m, s, lds_m[tid][w], lds_s[tid][w]);
        float lse = m + __logf(s);
        nll[r0 + tid] = lse - lds_lablogit[tid];
    }
}

// ---------------------------------------------------------------------------
// Kernel 2 (fallback, small workspace): on-the-fly fp32 -> bf16 hi/lo split.
// ---------------------------------------------------------------------------
__global__ __launch_bounds__(256) void closs_main_fly(
        const float* __restrict__ feat, const float* __restrict__ feat2,
        const int*   __restrict__ labels, const float* __restrict__ temp,
        const float* __restrict__ x2, const float* __restrict__ y2,
        float* __restrict__ nll) {
    __shared__ float lds_x2[16];
    __shared__ float lds_lablogit[16];
    __shared__ int   lds_lab[16];
    __shared__ float lds_m[16][8];
    __shared__ float lds_s[16][8];

    const int tid  = threadIdx.x;
    const int lane = tid & 31;
    const int wave = tid >> 5;
    const int hl   = lane >> 4;
    const int lid  = lane & 15;
    const int r0   = blockIdx.x * 16;

    if (tid < 16) {
        lds_x2[tid]       = x2[r0 + tid];
        lds_lab[tid]      = labels[r0 + tid];
        lds_lablogit[tid] = 0.0f;
    }
    __syncthreads();

    const float inv_temp = 1.0f / temp[0];
    const float* abase = feat + (size_t)(r0 + lid) * Dd + hl * 16;

    float mrun[8], srun[8];
    #pragma unroll
    for (int v = 0; v < 8; ++v) { mrun[v] = -INFINITY; srun[v] = 0.0f; }

    for (int tile = wave; tile < Mm / 16; tile += 8) {
        const int c0 = tile * 16;
        const float* bbase = feat2 + (size_t)(c0 + lid) * Dd + hl * 16;

        v8f acc0 = {}, acc1 = {}, acc2 = {};
        for (int k0 = 0; k0 < Dd; k0 += 32) {
            float av[16], bv[16];
            *(float4*)(av + 0)  = *(const float4*)(abase + k0 + 0);
            *(float4*)(av + 4)  = *(const float4*)(abase + k0 + 4);
            *(float4*)(av + 8)  = *(const float4*)(abase + k0 + 8);
            *(float4*)(av + 12) = *(const float4*)(abase + k0 + 12);
            *(float4*)(bv + 0)  = *(const float4*)(bbase + k0 + 0);
            *(float4*)(bv + 4)  = *(const float4*)(bbase + k0 + 4);
            *(float4*)(bv + 8)  = *(const float4*)(bbase + k0 + 8);
            *(float4*)(bv + 12) = *(const float4*)(bbase + k0 + 12);

            v16bf ahi, alo, bhi, blo;
            #pragma unroll
            for (int i = 0; i < 16; ++i) {
                __bf16 ha = (__bf16)av[i];
                ahi[i] = ha;
                alo[i] = (__bf16)(av[i] - (float)ha);
                __bf16 hb = (__bf16)bv[i];
                bhi[i] = hb;
                blo[i] = (__bf16)(bv[i] - (float)hb);
            }
            acc0 = __builtin_amdgcn_wmma_f32_16x16x32_bf16(false, ahi, false, bhi,
                                                           (short)0, acc0, false, false);
            acc1 = __builtin_amdgcn_wmma_f32_16x16x32_bf16(false, ahi, false, blo,
                                                           (short)0, acc1, false, false);
            acc2 = __builtin_amdgcn_wmma_f32_16x16x32_bf16(false, alo, false, bhi,
                                                           (short)0, acc2, false, false);
        }

        const float y2c = y2[c0 + lid];
        #pragma unroll
        for (int v = 0; v < 8; ++v) {
            const int row = v + 8 * hl;
            float dot  = acc0[v] + (acc1[v] + acc2[v]);
            float d2   = lds_x2[row] + y2c - 2.0f * dot;
            float dist = __builtin_amdgcn_sqrtf(fmaxf(d2, 0.0f));
            float lg   = -dist * inv_temp;
            if (lds_lab[row] == c0 + lid) lds_lablogit[row] = lg;
            lse_update(mrun[v], srun[v], lg);
        }
    }

    #pragma unroll
    for (int off = 1; off < 16; off <<= 1) {
        #pragma unroll
        for (int v = 0; v < 8; ++v) {
            float m2 = __shfl_xor(mrun[v], off, 32);
            float s2 = __shfl_xor(srun[v], off, 32);
            lse_merge(mrun[v], srun[v], m2, s2);
        }
    }
    if (lid == 0) {
        #pragma unroll
        for (int v = 0; v < 8; ++v) {
            lds_m[v + 8 * hl][wave] = mrun[v];
            lds_s[v + 8 * hl][wave] = srun[v];
        }
    }
    __syncthreads();

    if (tid < 16) {
        float m = lds_m[tid][0];
        float s = lds_s[tid][0];
        #pragma unroll
        for (int w = 1; w < 8; ++w)
            lse_merge(m, s, lds_m[tid][w], lds_s[tid][w]);
        float lse = m + __logf(s);
        nll[r0 + tid] = lse - lds_lablogit[tid];
    }
}

// ---------------------------------------------------------------------------
// Kernel 3: mean of per-row NLL -> scalar output.
// ---------------------------------------------------------------------------
__global__ __launch_bounds__(256) void closs_reduce(const float* __restrict__ nll,
                                                    float* __restrict__ out) {
    __shared__ float sm[256];
    float s = 0.0f;
    for (int i = threadIdx.x; i < Nn; i += 256) s += nll[i];
    sm[threadIdx.x] = s;
    __syncthreads();
    for (int off = 128; off >= 1; off >>= 1) {
        if ((int)threadIdx.x < off) sm[threadIdx.x] += sm[threadIdx.x + off];
        __syncthreads();
    }
    if (threadIdx.x == 0) out[0] = sm[0] * (1.0f / (float)Nn);
}

extern "C" void kernel_launch(void* const* d_in, const int* in_sizes, int n_in,
                              void* d_out, int out_size, void* d_ws, size_t ws_size,
                              hipStream_t stream) {
    const float* feat   = (const float*)d_in[0];
    const float* feat2  = (const float*)d_in[1];
    const int*   labels = (const int*)d_in[2];
    const float* temp   = (const float*)d_in[3];
    float* out = (float*)d_out;

    float* x2  = (float*)d_ws;       // [Nn]
    float* y2  = x2 + Nn;            // [Mm]
    float* nll = y2 + Mm;            // [Nn]
    const size_t head = (size_t)(Nn + Mm + Nn) * sizeof(float);   // 48 KB

    const size_t elems = (size_t)Nn * Dd;                 // feat & feat2 same size
    const size_t need  = head + 4 * elems * sizeof(unsigned short);  // +16 MB

    closs_norms<<<(Nn + Mm) / 8, 256, 0, stream>>>(feat, feat2, x2, y2);

    if (ws_size >= need) {
        unsigned short* fhi = (unsigned short*)((char*)d_ws + head);
        unsigned short* flo = fhi + elems;
        unsigned short* ghi = flo + elems;
        unsigned short* glo = ghi + elems;
        const int n4 = (int)(elems / 4);
        closs_split<<<(n4 + 255) / 256, 256, 0, stream>>>(feat,  fhi, flo, n4);
        closs_split<<<(n4 + 255) / 256, 256, 0, stream>>>(feat2, ghi, glo, n4);
        closs_main_pre<<<Nn / 16, 256, 0, stream>>>(fhi, flo, ghi, glo,
                                                    labels, temp, x2, y2, nll);
    } else {
        closs_main_fly<<<Nn / 16, 256, 0, stream>>>(feat, feat2,
                                                    labels, temp, x2, y2, nll);
    }
    closs_reduce<<<1, 256, 0, stream>>>(nll, out);
}
